// MultiHeadAttention_53386443489374
// MI455X (gfx1250) — matrix operand order repaired and checked by reading, not compiled, not run on hardware
//
#include <hip/hip_runtime.h>
#include <stdint.h>

#define D_MODEL   1024
#define N_HEADS   16
#define DK        64
#define SEQ       2048
#define NB        4
#define ROWS      (NB * SEQ)          // 8192
#define MASK_FILL_F (-1e9f)

typedef __bf16 bf16_t;
typedef __attribute__((ext_vector_type(16))) bf16_t v16bf;
typedef __attribute__((ext_vector_type(8)))  bf16_t v8bf;
typedef __attribute__((ext_vector_type(8)))  float  v8f;
typedef __attribute__((ext_vector_type(4)))  float  v4f;

// -------------------- WMMA core --------------------

static __device__ __forceinline__ v8f wmma_bf16(v16bf a, v16bf b, v8f c) {
  // D = A(16x32 bf16) * B(32x16 bf16) + C(16x16 f32)
  return __builtin_amdgcn_wmma_f32_16x16x32_bf16(false, a, false, b, (short)0, c,
                                                 false, false);
}

// A fragment (16x32 bf16) from row-major A[*, lda]; rows m..m+15, cols k0..k0+31.
// Lane l: row = l&15; kb = (l&16)?8:0; elems 0..7 <- K=k0+kb..+7, 8..15 <- K=k0+16+kb..+7
static __device__ __forceinline__ v16bf frag_a_bf16(const bf16_t* __restrict__ A,
                                                    int lda, int m, int k0) {
  const int l  = threadIdx.x & 31;
  const int kb = (l & 16) ? 8 : 0;
  const bf16_t* p = A + (size_t)(m + (l & 15)) * lda + (k0 + kb);
  v8bf lo = *(const v8bf*)(p);
  v8bf hi = *(const v8bf*)(p + 16);
  v16bf r;
#pragma unroll
  for (int i = 0; i < 8; ++i) { r[i] = lo[i]; r[8 + i] = hi[i]; }
  return r;
}

// Same A fragment but source is fp32 (converted to bf16 on the fly).
static __device__ __forceinline__ v16bf frag_a_f32(const float* __restrict__ A,
                                                   int lda, int m, int k0) {
  const int l  = threadIdx.x & 31;
  const int kb = (l & 16) ? 8 : 0;
  const float* p = A + (size_t)(m + (l & 15)) * lda + (k0 + kb);
  v4f a0 = *(const v4f*)(p);
  v4f a1 = *(const v4f*)(p + 4);
  v4f a2 = *(const v4f*)(p + 16);
  v4f a3 = *(const v4f*)(p + 20);
  v16bf r;
#pragma unroll
  for (int i = 0; i < 4; ++i) {
    r[i]      = (bf16_t)a0[i];
    r[4 + i]  = (bf16_t)a1[i];
    r[8 + i]  = (bf16_t)a2[i];
    r[12 + i] = (bf16_t)a3[i];
  }
  return r;
}

// B fragment (32x16 bf16) from B^T stored row-major [N, ldb] (K contiguous).
// Lane l: col n = l&15; kb = (l&16)?8:0; elems 0..7 <- K=k0+kb..+7, 8..15 <- K=k0+16+kb..+7
static __device__ __forceinline__ v16bf frag_b_bt(const bf16_t* __restrict__ BT,
                                                  int ldb, int n, int k0) {
  const int l  = threadIdx.x & 31;
  const int kb = (l & 16) ? 8 : 0;
  const bf16_t* p = BT + (size_t)(n + (l & 15)) * ldb + (k0 + kb);
  v8bf lo = *(const v8bf*)(p);
  v8bf hi = *(const v8bf*)(p + 16);
  v16bf r;
#pragma unroll
  for (int i = 0; i < 8; ++i) { r[i] = lo[i]; r[8 + i] = hi[i]; }
  return r;
}

// -------------------- generic WMMA GEMM --------------------
// C[z] = A[z] (M x K, row-major) * B[z] where BT is B^T stored [N, K] row-major.
// Block = 256 threads = 8 waves arranged 4(M) x 2(N).
// Block tile 256(M) x 64(N); wave tile 64 x 32 -> 8 accumulators, 8 WMMA / K-step.
// C base offset per z: (z/zdiv)*strideC1 + (z%zdiv)*strideC2   (element units)

template <bool A_F32, bool OUT_BF16>
__global__ __launch_bounds__(256) void gemm_wmma_kernel(
    const void* __restrict__ Av, int lda, long long strideA,
    const bf16_t* __restrict__ BT, int ldb, long long strideB,
    void* __restrict__ Cv, int ldc, int zdiv, long long strideC1,
    long long strideC2, int Kdim) {
  const int z  = blockIdx.z;
  const int w  = threadIdx.x >> 5;
  const int m0 = blockIdx.y * 256 + (w & 3) * 64;
  const int n0 = blockIdx.x * 64 + (w >> 2) * 32;
  const bf16_t* Bz = BT + (size_t)z * strideB;

  const v8f vzero = {0.f, 0.f, 0.f, 0.f, 0.f, 0.f, 0.f, 0.f};
  v8f acc[4][2] = {{vzero, vzero}, {vzero, vzero}, {vzero, vzero}, {vzero, vzero}};

  for (int k0 = 0; k0 < Kdim; k0 += 32) {
    if (k0 + 32 < Kdim) {  // prefetch next K-slab of B (global_prefetch_b8)
      __builtin_prefetch(
          (const void*)(Bz + (size_t)(n0 + (threadIdx.x & 31)) * ldb + k0 + 32),
          0, 1);
    }
    v16bf a[4];
    if constexpr (A_F32) {
      const float* A = (const float*)Av + (size_t)z * strideA;
#pragma unroll
      for (int im = 0; im < 4; ++im) a[im] = frag_a_f32(A, lda, m0 + im * 16, k0);
    } else {
      const bf16_t* A = (const bf16_t*)Av + (size_t)z * strideA;
#pragma unroll
      for (int im = 0; im < 4; ++im) a[im] = frag_a_bf16(A, lda, m0 + im * 16, k0);
    }
    v16bf b0 = frag_b_bt(Bz, ldb, n0, k0);
    v16bf b1 = frag_b_bt(Bz, ldb, n0 + 16, k0);
#pragma unroll
    for (int im = 0; im < 4; ++im) {
      acc[im][0] = wmma_bf16(a[im], b0, acc[im][0]);
      acc[im][1] = wmma_bf16(a[im], b1, acc[im][1]);
    }
  }

  const int l  = threadIdx.x & 31;
  const int nn = l & 15;
  const int mb = (l & 16) ? 8 : 0;
  const size_t cbase =
      (size_t)(z / zdiv) * strideC1 + (size_t)(z % zdiv) * strideC2;
#pragma unroll
  for (int im = 0; im < 4; ++im)
#pragma unroll
    for (int in = 0; in < 2; ++in)
#pragma unroll
      for (int v = 0; v < 8; ++v) {
        const int m = m0 + im * 16 + mb + v;
        const int n = n0 + in * 16 + nn;
        const size_t idx = cbase + (size_t)m * ldc + n;
        if constexpr (OUT_BF16)
          ((bf16_t*)Cv)[idx] = (bf16_t)acc[im][in][v];
        else
          ((float*)Cv)[idx] = acc[im][in][v];
      }
}

// -------------------- V projection with transposed output --------------------
// CT layout: [NB][D_MODEL][SEQ]; global row m -> (b = m>>11, s = m&2047), col n.

__global__ __launch_bounds__(256) void gemm_wmma_outT_kernel(
    const float* __restrict__ A, int lda, const bf16_t* __restrict__ BT,
    int ldb, bf16_t* __restrict__ CT, int Kdim) {
  const int w  = threadIdx.x >> 5;
  const int m0 = blockIdx.y * 256 + (w & 3) * 64;
  const int n0 = blockIdx.x * 64 + (w >> 2) * 32;

  const v8f vzero = {0.f, 0.f, 0.f, 0.f, 0.f, 0.f, 0.f, 0.f};
  v8f acc[4][2] = {{vzero, vzero}, {vzero, vzero}, {vzero, vzero}, {vzero, vzero}};

  for (int k0 = 0; k0 < Kdim; k0 += 32) {
    v16bf a[4];
#pragma unroll
    for (int im = 0; im < 4; ++im) a[im] = frag_a_f32(A, lda, m0 + im * 16, k0);
    v16bf b0 = frag_b_bt(BT, ldb, n0, k0);
    v16bf b1 = frag_b_bt(BT, ldb, n0 + 16, k0);
#pragma unroll
    for (int im = 0; im < 4; ++im) {
      acc[im][0] = wmma_bf16(a[im], b0, acc[im][0]);
      acc[im][1] = wmma_bf16(a[im], b1, acc[im][1]);
    }
  }

  const int l  = threadIdx.x & 31;
  const int nn = l & 15;
  const int mb = (l & 16) ? 8 : 0;
#pragma unroll
  for (int im = 0; im < 4; ++im)
#pragma unroll
    for (int in = 0; in < 2; ++in)
#pragma unroll
      for (int v = 0; v < 8; ++v) {
        const int m = m0 + im * 16 + mb + v;
        const int n = n0 + in * 16 + nn;
        const int b = m >> 11;
        const int s = m & (SEQ - 1);
        CT[((size_t)b * D_MODEL + n) * SEQ + s] = (bf16_t)acc[im][in][v];
      }
}

// -------------------- scores = (Q K^T)/sqrt(dk), mask, NT fp32 out --------------------

__global__ __launch_bounds__(256) void scores_wmma_kernel(
    const bf16_t* __restrict__ Qp, const bf16_t* __restrict__ Kp,
    const uint8_t* __restrict__ mask, float* __restrict__ attn) {
  const int z = blockIdx.z;
  const int b = z >> 4;
  const int h = z & 15;
  const bf16_t* Qb = Qp + (size_t)b * SEQ * D_MODEL + h * DK;
  const bf16_t* Kb = Kp + (size_t)b * SEQ * D_MODEL + h * DK;

  const int w  = threadIdx.x >> 5;
  const int m0 = blockIdx.y * 256 + (w & 3) * 64;
  const int n0 = blockIdx.x * 64 + (w >> 2) * 32;

  const v8f vzero = {0.f, 0.f, 0.f, 0.f, 0.f, 0.f, 0.f, 0.f};
  v8f acc[4][2] = {{vzero, vzero}, {vzero, vzero}, {vzero, vzero}, {vzero, vzero}};

#pragma unroll
  for (int k0 = 0; k0 < DK; k0 += 32) {
    v16bf a[4];
#pragma unroll
    for (int im = 0; im < 4; ++im) a[im] = frag_a_bf16(Qb, D_MODEL, m0 + im * 16, k0);
    v16bf b0 = frag_b_bt(Kb, D_MODEL, n0, k0);
    v16bf b1 = frag_b_bt(Kb, D_MODEL, n0 + 16, k0);
#pragma unroll
    for (int im = 0; im < 4; ++im) {
      acc[im][0] = wmma_bf16(a[im], b0, acc[im][0]);
      acc[im][1] = wmma_bf16(a[im], b1, acc[im][1]);
    }
  }

  const int l  = threadIdx.x & 31;
  const int nn = l & 15;
  const int mb = (l & 16) ? 8 : 0;
  const uint8_t* mrow = mask + (size_t)b * SEQ * SEQ;
  float* orow = attn + (size_t)z * SEQ * SEQ;
  const float scale = 0.125f;  // 1/sqrt(64)
#pragma unroll
  for (int im = 0; im < 4; ++im)
#pragma unroll
    for (int in = 0; in < 2; ++in)
#pragma unroll
      for (int v = 0; v < 8; ++v) {
        const int m = m0 + im * 16 + mb + v;
        const int n = n0 + in * 16 + nn;
        float val = acc[im][in][v] * scale;
        if (mrow[(size_t)m * SEQ + n]) val = MASK_FILL_F;
        // 1GB attn stream >> 192MB L2: keep it out of cache (TH=NT).
        __builtin_nontemporal_store(val, &orow[(size_t)m * SEQ + n]);
      }
}

// -------------------- row softmax in place (one block per row) --------------------

__global__ __launch_bounds__(256) void softmax_kernel(float* __restrict__ attn) {
  __shared__ float red[256];
  float* row = attn + (size_t)blockIdx.x * SEQ;
  const int t = threadIdx.x;
  float x[8];
  float lmax = -3.0e38f;
#pragma unroll
  for (int i = 0; i < 8; ++i) {
    x[i] = __builtin_nontemporal_load(&row[t + i * 256]);
    lmax = fmaxf(lmax, x[i]);
  }
  red[t] = lmax;
  __syncthreads();
  for (int s = 128; s > 0; s >>= 1) {
    if (t < s) red[t] = fmaxf(red[t], red[t + s]);
    __syncthreads();
  }
  const float mx = red[0];
  __syncthreads();
  float lsum = 0.f;
#pragma unroll
  for (int i = 0; i < 8; ++i) {
    x[i] = __expf(x[i] - mx);
    lsum += x[i];
  }
  red[t] = lsum;
  __syncthreads();
  for (int s = 128; s > 0; s >>= 1) {
    if (t < s) red[t] += red[t + s];
    __syncthreads();
  }
  const float inv = 1.0f / red[0];
#pragma unroll
  for (int i = 0; i < 8; ++i)
    __builtin_nontemporal_store(x[i] * inv, &row[t + i * 256]);
}

// -------------------- residual + LayerNorm (one block per row) --------------------

__global__ __launch_bounds__(256) void residual_ln_kernel(
    const float* __restrict__ fc, const float* __restrict__ resid,
    const float* __restrict__ gamma, const float* __restrict__ beta,
    float* __restrict__ out) {
  __shared__ float rs[256];
  __shared__ float rq[256];
  const int r = blockIdx.x, t = threadIdx.x;
  const size_t base = (size_t)r * D_MODEL;
  float x[4];
  float s = 0.f, q = 0.f;
#pragma unroll
  for (int i = 0; i < 4; ++i) {
    const int c = t + i * 256;
    x[i] = fc[base + c] + resid[base + c];
    s += x[i];
    q += x[i] * x[i];
  }
  rs[t] = s;
  rq[t] = q;
  __syncthreads();
  for (int st = 128; st > 0; st >>= 1) {
    if (t < st) {
      rs[t] += rs[t + st];
      rq[t] += rq[t + st];
    }
    __syncthreads();
  }
  const float mu  = rs[0] * (1.0f / D_MODEL);
  const float var = rq[0] * (1.0f / D_MODEL) - mu * mu;
  const float inv = rsqrtf(var + 1e-5f);
#pragma unroll
  for (int i = 0; i < 4; ++i) {
    const int c = t + i * 256;
    out[base + c] = (x[i] - mu) * inv * gamma[c] + beta[c];
  }
}

// -------------------- weight transpose+convert: W[K][N] f32 -> WT[N][K] bf16 --------

__global__ void transpose_w_kernel(const float* __restrict__ W,
                                   bf16_t* __restrict__ WT) {
  const int n = blockIdx.x * 16 + threadIdx.x;
  const int k = blockIdx.y * 16 + threadIdx.y;
  WT[(size_t)n * D_MODEL + k] = (bf16_t)W[(size_t)k * D_MODEL + n];
}

// -------------------- host launcher --------------------

extern "C" void kernel_launch(void* const* d_in, const int* in_sizes, int n_in,
                              void* d_out, int out_size, void* d_ws,
                              size_t ws_size, hipStream_t stream) {
  (void)in_sizes; (void)n_in; (void)out_size; (void)ws_size;

  const float*   inQ   = (const float*)d_in[0];
  const float*   inK   = (const float*)d_in[1];
  const float*   inV   = (const float*)d_in[2];
  const uint8_t* mask  = (const uint8_t*)d_in[3];
  const float*   Wq    = (const float*)d_in[4];
  const float*   Wk    = (const float*)d_in[5];
  const float*   Wv    = (const float*)d_in[6];
  const float*   Wfc   = (const float*)d_in[7];
  const float*   gamma = (const float*)d_in[8];
  const float*   beta  = (const float*)d_in[9];

  float* outLN = (float*)d_out;                               // [B,S,D_MODEL]
  float* attn  = (float*)d_out + (size_t)ROWS * D_MODEL;      // [B,H,S,S]

  char* ws = (char*)d_ws;
  size_t off = 0;
  auto carve = [&](size_t bytes) -> void* {
    void* p = (void*)(ws + off);
    off += (bytes + 255) & ~(size_t)255;
    return p;
  };
  bf16_t* WqT  = (bf16_t*)carve((size_t)D_MODEL * D_MODEL * 2);
  bf16_t* WkT  = (bf16_t*)carve((size_t)D_MODEL * D_MODEL * 2);
  bf16_t* WvT  = (bf16_t*)carve((size_t)D_MODEL * D_MODEL * 2);
  bf16_t* WfcT = (bf16_t*)carve((size_t)D_MODEL * D_MODEL * 2);
  bf16_t* Qp   = (bf16_t*)carve((size_t)ROWS * D_MODEL * 2);  // [B*S, H*DK]
  bf16_t* Kp   = (bf16_t*)carve((size_t)ROWS * D_MODEL * 2);  // [B*S, H*DK]
  bf16_t* VpT  = (bf16_t*)carve((size_t)ROWS * D_MODEL * 2);  // [B][H*DK][S]
  bf16_t* ctx  = (bf16_t*)carve((size_t)ROWS * D_MODEL * 2);  // [B*S, H*DV]
  float*  fcb  = (float*)carve((size_t)ROWS * D_MODEL * 4);   // [B*S, D_MODEL]

  const dim3 tb(256, 1, 1);

  // 1) weights -> bf16, transposed to [N][K]
  {
    const dim3 tw(16, 16, 1);
    const dim3 gw(D_MODEL / 16, D_MODEL / 16, 1);
    transpose_w_kernel<<<gw, tw, 0, stream>>>(Wq, WqT);
    transpose_w_kernel<<<gw, tw, 0, stream>>>(Wk, WkT);
    transpose_w_kernel<<<gw, tw, 0, stream>>>(Wv, WvT);
    transpose_w_kernel<<<gw, tw, 0, stream>>>(Wfc, WfcT);
  }

  // 2) projections (A = fp32 inputs, converted in-register)
  const dim3 gproj(D_MODEL / 64, ROWS / 256, 1);  // (16, 32)
  gemm_wmma_kernel<true, true><<<gproj, tb, 0, stream>>>(
      inQ, D_MODEL, 0, WqT, D_MODEL, 0, Qp, D_MODEL, 1, 0, 0, D_MODEL);
  gemm_wmma_kernel<true, true><<<gproj, tb, 0, stream>>>(
      inK, D_MODEL, 0, WkT, D_MODEL, 0, Kp, D_MODEL, 1, 0, 0, D_MODEL);
  gemm_wmma_outT_kernel<<<gproj, tb, 0, stream>>>(inV, D_MODEL, WvT, D_MODEL,
                                                  VpT, D_MODEL);

  // 3) scores = QK^T/sqrt(dk) with mask -> attn region of d_out (fp32, NT)
  {
    const dim3 gsc(SEQ / 64, SEQ / 256, NB * N_HEADS);  // (32, 8, 64)
    scores_wmma_kernel<<<gsc, tb, 0, stream>>>(Qp, Kp, mask, attn);
  }

  // 4) softmax over each of B*H*S rows, in place
  softmax_kernel<<<dim3(NB * N_HEADS * SEQ, 1, 1), tb, 0, stream>>>(attn);

  // 5) context = attn @ V  (A = fp32 attn converted in-register; B = V^T)
  {
    const dim3 gctx(1, SEQ / 256, NB * N_HEADS);  // (1, 8, 64), N = 64 per head
    gemm_wmma_kernel<true, true><<<gctx, tb, 0, stream>>>(
        attn, SEQ, (long long)SEQ * SEQ, VpT, SEQ, (long long)DK * SEQ, ctx,
        D_MODEL, N_HEADS, (long long)SEQ * D_MODEL, (long long)DK, SEQ);
  }

  // 6) fc = context @ W_fc -> fp32
  gemm_wmma_kernel<false, false><<<gproj, tb, 0, stream>>>(
      ctx, D_MODEL, 0, WfcT, D_MODEL, 0, fcb, D_MODEL, 1, 0, 0, D_MODEL);

  // 7) layernorm(fc + input_Q) -> first region of d_out
  residual_ln_kernel<<<dim3(ROWS, 1, 1), tb, 0, stream>>>(fcb, inQ, gamma,
                                                          beta, outLN);
}
